// HoloKVSimulatorLayer_76716705841479
// MI455X (gfx1250) — matrix operand order repaired and checked by reading, not compiled, not run on hardware
//
#include <hip/hip_runtime.h>
#include <hip/hip_bf16.h>
#include <stdint.h>

// ---------------------------------------------------------------------------
// HoloKV attention layer for gfx1250 (MI455X).
// B=2, S=2048, H=1024, NH=NKV=16, HD=64, R=16, KF=4, theta=1e6
// Matmuls via v_wmma_f32_16x16x32_f16; GEMM staged through LDS with
// async global->LDS copies (global_load_async_to_lds_b128, ASYNCcnt).
// ---------------------------------------------------------------------------

#define B_   2
#define S_   2048
#define H_   1024
#define NH_  16
#define HD_  64
#define R_   16
#define KF_  4
#define M_   (B_ * S_)          // 4096
#define BH_  (B_ * NH_)         // 32

#define USE_ASYNC_LDS 1

typedef __attribute__((ext_vector_type(16))) _Float16 v16h;
typedef __attribute__((ext_vector_type(8)))  _Float16 v8h;
typedef __attribute__((ext_vector_type(8)))  float    v8f;

__device__ __forceinline__ v8f wmma_f16(v16h a, v16h b, v8f c) {
  return __builtin_amdgcn_wmma_f32_16x16x32_f16(false, a, false, b, (short)0, c,
                                                false, false);
}

#if USE_ASYNC_LDS
// Async global->LDS 16B copy (per lane). vdst = LDS byte address (low 32 bits
// of the generic pointer per the ISA aperture rule), vaddr = 64-bit global.
__device__ __forceinline__ void async_load_b128(const void* gptr, void* lptr) {
  uint32_t loff = (uint32_t)(uintptr_t)lptr;
  unsigned long long gaddr = (unsigned long long)(uintptr_t)gptr;
  asm volatile("global_load_async_to_lds_b128 %0, %1, off"
               :
               : "v"(loff), "v"(gaddr)
               : "memory");
}
#endif

__device__ __forceinline__ void wait_async0() {
#if defined(__has_builtin) && __has_builtin(__builtin_amdgcn_s_wait_asynccnt)
  __builtin_amdgcn_s_wait_asynccnt(0);
#else
  asm volatile("s_wait_asynccnt 0" ::: "memory");
#endif
}

// ---------------------------------------------------------------------------
__global__ void conv_f32_to_f16(const float* __restrict__ in,
                                _Float16* __restrict__ out, int n) {
  int i = blockIdx.x * blockDim.x + threadIdx.x;
  if (i < n) out[i] = (_Float16)in[i];
}

// ---------------------------------------------------------------------------
// LDS-tiled WMMA GEMM: C[M,N] = A[M,K] * W[N,K]^T, f32 accumulate.
// Block = 256 threads (8 waves). Block tile 128(M) x 64(N), K-step 32,
// double-buffered LDS. Wave wv computes rows [wv*16, wv*16+16) x all 64 cols.
// All LDS fragment loads are issued before the WMMA burst so only one
// dscnt wait is needed per K-step.
// ---------------------------------------------------------------------------
#define BM_ 128
#define BN_ 64
#define BK_ 32
#define LPAD_ 40   // padded LDS row length (halfs); 80B rows keep 16B align

__global__ __launch_bounds__(256) void gemm_lds_wmma(
    const _Float16* __restrict__ A, const _Float16* __restrict__ W,
    float* __restrict__ C, int M, int N, int K) {
  __shared__ _Float16 As[2][BM_][LPAD_];
  __shared__ _Float16 Bs[2][BN_][LPAD_];

  const int tid  = threadIdx.x;
  const int wv   = tid >> 5;
  const int lane = tid & 31;
  const int hlf  = lane >> 4;
  const int l16  = lane & 15;
  const int m0   = blockIdx.y * BM_;
  const int n0   = blockIdx.x * BN_;

  // chunk maps (16B chunks): A tile = 512 chunks (2/thread), B tile = 256.
  const int ar0 = tid >> 2, as0 = (tid & 3);             // A chunk j=0
  const int ar1 = (tid + 256) >> 2, as1 = (tid & 3);     // A chunk j=1
  const int brw = tid >> 2, bsb = (tid & 3);             // B chunk

  const _Float16* gA0 = A + (size_t)(m0 + ar0) * K + as0 * 8;
  const _Float16* gA1 = A + (size_t)(m0 + ar1) * K + as1 * 8;
  const _Float16* gB  = W + (size_t)(n0 + brw) * K + bsb * 8;

  v8f acc[4];
#pragma unroll
  for (int t = 0; t < 4; ++t) acc[t] = (v8f){};

  const int ksteps = K / BK_;
  int buf = 0;

#if USE_ASYNC_LDS
  async_load_b128(gA0, &As[0][ar0][as0 * 8]);
  async_load_b128(gA1, &As[0][ar1][as1 * 8]);
  async_load_b128(gB,  &Bs[0][brw][bsb * 8]);
  wait_async0();
  __syncthreads();
#else
  {
    v8h ta0 = *(const v8h*)gA0;
    v8h ta1 = *(const v8h*)gA1;
    v8h tb  = *(const v8h*)gB;
    *(v8h*)&As[0][ar0][as0 * 8] = ta0;
    *(v8h*)&As[0][ar1][as1 * 8] = ta1;
    *(v8h*)&Bs[0][brw][bsb * 8] = tb;
  }
  __syncthreads();
#endif

  for (int ks = 0; ks < ksteps; ++ks) {
    const int has_next = (ks + 1) < ksteps;     // block-uniform

    if ((ks + 2) < ksteps) {                    // L2 prefetch ahead
      __builtin_prefetch(gA0 + (size_t)(ks + 2) * BK_, 0, 1);
      __builtin_prefetch(gB + (size_t)(ks + 2) * BK_, 0, 1);
    }

#if USE_ASYNC_LDS
    if (has_next) {
      const size_t ko = (size_t)(ks + 1) * BK_;
      async_load_b128(gA0 + ko, &As[buf ^ 1][ar0][as0 * 8]);
      async_load_b128(gA1 + ko, &As[buf ^ 1][ar1][as1 * 8]);
      async_load_b128(gB + ko,  &Bs[buf ^ 1][brw][bsb * 8]);
    }
#else
    v8h na0, na1, nb;
    if (has_next) {                             // start loads early
      const size_t ko = (size_t)(ks + 1) * BK_;
      na0 = *(const v8h*)(gA0 + ko);
      na1 = *(const v8h*)(gA1 + ko);
      nb  = *(const v8h*)(gB + ko);
    }
#endif

    // ---- compute on current buffer: preload ALL fragments, then WMMA burst
    {
      const _Float16* arow = &As[buf][wv * 16 + l16][0];
      v8h a0 = *(const v8h*)(arow + 8 * hlf);
      v8h a1 = *(const v8h*)(arow + 16 + 8 * hlf);

      v8h bl[4], bh4[4];
#pragma unroll
      for (int nt = 0; nt < 4; ++nt) {
        const _Float16* brow = &Bs[buf][nt * 16 + l16][0];
        bl[nt]  = *(const v8h*)(brow + 8 * hlf);
        bh4[nt] = *(const v8h*)(brow + 16 + 8 * hlf);
      }

      v16h a;
#pragma unroll
      for (int e = 0; e < 8; ++e) { a[e] = a0[e]; a[e + 8] = a1[e]; }
      v16h bfr[4];
#pragma unroll
      for (int nt = 0; nt < 4; ++nt) {
#pragma unroll
        for (int e = 0; e < 8; ++e) {
          bfr[nt][e] = bl[nt][e];
          bfr[nt][e + 8] = bh4[nt][e];
        }
      }
#pragma unroll
      for (int nt = 0; nt < 4; ++nt)
        acc[nt] = wmma_f16(a, bfr[nt], acc[nt]);
    }

#if USE_ASYNC_LDS
    wait_async0();
#else
    if (has_next) {
      *(v8h*)&As[buf ^ 1][ar0][as0 * 8] = na0;
      *(v8h*)&As[buf ^ 1][ar1][as1 * 8] = na1;
      *(v8h*)&Bs[buf ^ 1][brw][bsb * 8] = nb;
    }
#endif
    __syncthreads();
    buf ^= 1;
  }

  // ---- epilogue: D layout -> C ----
#pragma unroll
  for (int nt = 0; nt < 4; ++nt) {
#pragma unroll
    for (int r = 0; r < 8; ++r) {
      const int row = m0 + wv * 16 + r + 8 * hlf;
      const int col = n0 + nt * 16 + l16;
      C[(size_t)row * N + col] = acc[nt][r];
    }
  }
}

// ---------------------------------------------------------------------------
// LoRA down-projection: T[m,r] = sum_k X[m,k] * Amat[r,k]  (rank 16, f32)
// ---------------------------------------------------------------------------
__global__ void lora_down(const float* __restrict__ X,
                          const float* __restrict__ Amat,
                          float* __restrict__ T) {
  int idx = blockIdx.x * blockDim.x + threadIdx.x;
  if (idx >= M_ * R_) return;
  int r = idx & (R_ - 1);
  int m = idx >> 4;
  const float* xr = X + (size_t)m * H_;
  const float* arow = Amat + (size_t)r * H_;
  float acc = 0.f;
  for (int k = 0; k < H_; ++k) acc += xr[k] * arow[k];
  T[idx] = acc;
}

// ---------------------------------------------------------------------------
__global__ void add_bias_lora(float* __restrict__ Y,
                              const float* __restrict__ bias,
                              const float* __restrict__ T,
                              const float* __restrict__ Bmat, int use_lora) {
  int idx = blockIdx.x * blockDim.x + threadIdx.x;
  if (idx >= M_ * H_) return;
  int n = idx & (H_ - 1);
  int m = idx >> 10;
  float acc = Y[idx] + bias[n];
  if (use_lora) {
    const float* tr = T + (size_t)m * R_;
    const float* brw = Bmat + (size_t)n * R_;
#pragma unroll
    for (int r = 0; r < R_; ++r) acc += tr[r] * brw[r];
  }
  Y[idx] = acc;
}

// ---------------------------------------------------------------------------
// RoPE + [B,S,H] -> [B,NH,S,HD]; q out f16, k/v out f32.
// ---------------------------------------------------------------------------
__global__ void rope_reorder(const float* __restrict__ qf,
                             const float* __restrict__ kf,
                             const float* __restrict__ vf,
                             const int* __restrict__ pos,
                             _Float16* __restrict__ qr,
                             float* __restrict__ kr,
                             float* __restrict__ vr) {
  int idx = blockIdx.x * blockDim.x + threadIdx.x;
  if (idx >= B_ * NH_ * S_ * HD_) return;
  int d = idx & 63;
  int s = (idx >> 6) & (S_ - 1);
  int h = (idx >> 17) & (NH_ - 1);
  int b = idx >> 21;
  int rowi = b * S_ + s;
  int coli = h * HD_ + d;
  float p = (float)pos[rowi];
  int fi = d & 31;
  float inv = __powf(1.0e6f, -((float)(2 * fi)) * (1.0f / 64.0f));
  float ang = p * inv;
  float cv = __cosf(ang), sv = __sinf(ang);
  size_t ri = (size_t)rowi * H_;
  float qv = qf[ri + coli];
  float kv = kf[ri + coli];
  int colo = (d < 32) ? coli + 32 : coli - 32;
  float sgn = (d < 32) ? -1.f : 1.f;
  float qo = qf[ri + colo];
  float ko = kf[ri + colo];
  qr[idx] = (_Float16)(qv * cv + sgn * qo * sv);
  kr[idx] = kv * cv + sgn * ko * sv;
  vr[idx] = vf[ri + coli];
}

// ---------------------------------------------------------------------------
__global__ void holo_bind(const float* __restrict__ kr,
                          const float* __restrict__ vr,
                          const float* __restrict__ cdma,
                          _Float16* __restrict__ Kn,
                          _Float16* __restrict__ Vn) {
  int idx = blockIdx.x * blockDim.x + threadIdx.x;
  if (idx >= BH_ * (S_ / KF_) * HD_) return;
  int d = idx & 63;
  int slot = (idx >> 6) & (S_ / KF_ - 1);
  int bh = idx >> 15;
  size_t base = ((size_t)bh * S_ + slot * KF_) * HD_ + d;
  float sk = 0.f, sv = 0.f;
  float cd[KF_];
#pragma unroll
  for (int j = 0; j < KF_; ++j) {
    cd[j] = cdma[j * HD_ + d];
    sk += kr[base + (size_t)j * HD_] * cd[j];
    sv += vr[base + (size_t)j * HD_] * cd[j];
  }
#pragma unroll
  for (int j = 0; j < KF_; ++j) {
    Kn[base + (size_t)j * HD_] = (_Float16)(sk * cd[j]);
    Vn[base + (size_t)j * HD_] = (_Float16)(sv * cd[j]);
  }
}

// ---------------------------------------------------------------------------
// Causal flash attention, WMMA core (wave-uniform control flow).
// ---------------------------------------------------------------------------
__global__ __launch_bounds__(256) void attn_wmma(
    const _Float16* __restrict__ Q,
    const _Float16* __restrict__ Kn,
    const _Float16* __restrict__ Vn,
    float* __restrict__ O) {
  __shared__ _Float16 pstage[8][16][32];

  const int wv   = threadIdx.x >> 5;
  const int lane = threadIdx.x & 31;
  const int hlf  = lane >> 4;
  const int l16  = lane & 15;
  const int bh   = blockIdx.y;
  const int qt   = blockIdx.x * 8 + wv;
  const int q0   = qt * 16;

  const _Float16* qb = Q  + (size_t)bh * S_ * HD_;
  const _Float16* kb = Kn + (size_t)bh * S_ * HD_;
  const _Float16* vb = Vn + (size_t)bh * S_ * HD_;
  float* ob = O + (size_t)bh * S_ * HD_;

  v16h qa0, qa1;
  {
    const _Float16* qrow = qb + (size_t)(q0 + l16) * HD_;
    v8h x0 = *(const v8h*)(qrow + 8 * hlf);
    v8h x1 = *(const v8h*)(qrow + 16 + 8 * hlf);
    v8h x2 = *(const v8h*)(qrow + 32 + 8 * hlf);
    v8h x3 = *(const v8h*)(qrow + 48 + 8 * hlf);
#pragma unroll
    for (int e = 0; e < 8; ++e) {
      qa0[e] = x0[e]; qa0[e + 8] = x1[e];
      qa1[e] = x2[e]; qa1[e + 8] = x3[e];
    }
  }

  float mrow[8], lrow[8];
  v8f oacc[4];
#pragma unroll
  for (int r = 0; r < 8; ++r) { mrow[r] = -1e30f; lrow[r] = 0.f; }
#pragma unroll
  for (int t = 0; t < 4; ++t) oacc[t] = (v8f){};

  const float scale = 0.0625f;  // 1/(sqrt(64)*sqrt(4))
  const int nkt = qt + 1;

  for (int kt2 = 0; kt2 * 2 < nkt; ++kt2) {
    v8f sc[2];
    sc[0] = (v8f){}; sc[1] = (v8f){};
#pragma unroll
    for (int sub = 0; sub < 2; ++sub) {
      const int ks = kt2 * 2 + sub;
      const _Float16* krow = kb + (size_t)(ks * 16 + l16) * HD_;
      v8h c0 = *(const v8h*)(krow + 8 * hlf);
      v8h c1 = *(const v8h*)(krow + 16 + 8 * hlf);
      v8h c2 = *(const v8h*)(krow + 32 + 8 * hlf);
      v8h c3 = *(const v8h*)(krow + 48 + 8 * hlf);
      v16h bk0, bk1;
#pragma unroll
      for (int e = 0; e < 8; ++e) {
        bk0[e] = c0[e]; bk0[e + 8] = c1[e];
        bk1[e] = c2[e]; bk1[e + 8] = c3[e];
      }
      sc[sub] = wmma_f16(qa0, bk0, sc[sub]);
      sc[sub] = wmma_f16(qa1, bk1, sc[sub]);
      const int kidx = ks * 16 + l16;
#pragma unroll
      for (int r = 0; r < 8; ++r) {
        const int qidx = q0 + r + 8 * hlf;
        float sval = sc[sub][r] * scale;
        sc[sub][r] = (kidx > qidx) ? -1e30f : sval;
      }
    }

#pragma unroll
    for (int r = 0; r < 8; ++r) {
      float mloc = fmaxf(sc[0][r], sc[1][r]);
      mloc = fmaxf(mloc, __shfl_xor(mloc, 8));
      mloc = fmaxf(mloc, __shfl_xor(mloc, 4));
      mloc = fmaxf(mloc, __shfl_xor(mloc, 2));
      mloc = fmaxf(mloc, __shfl_xor(mloc, 1));
      float mnew  = fmaxf(mrow[r], mloc);
      float alpha = __expf(mrow[r] - mnew);
      float p0 = __expf(sc[0][r] - mnew);
      float p1 = __expf(sc[1][r] - mnew);
      float ls = p0 + p1;
      ls += __shfl_xor(ls, 8);
      ls += __shfl_xor(ls, 4);
      ls += __shfl_xor(ls, 2);
      ls += __shfl_xor(ls, 1);
      lrow[r] = lrow[r] * alpha + ls;
      mrow[r] = mnew;
#pragma unroll
      for (int t = 0; t < 4; ++t) oacc[t][r] *= alpha;
      const int m = r + 8 * hlf;
      pstage[wv][m][l16]      = (_Float16)p0;
      pstage[wv][m][16 + l16] = (_Float16)p1;
    }
    asm volatile("s_wait_dscnt 0" ::: "memory");

    v16h pa;
#pragma unroll
    for (int e = 0; e < 8; ++e) {
      pa[e]     = pstage[wv][l16][16 * hlf + e];
      pa[e + 8] = pstage[wv][l16][16 * hlf + 8 + e];
    }
    asm volatile("s_wait_dscnt 0" ::: "memory");

#pragma unroll
    for (int t = 0; t < 4; ++t) {
      v16h vbf;
#pragma unroll
      for (int e = 0; e < 16; ++e)
        vbf[e] = vb[(size_t)(kt2 * 32 + e + 16 * hlf) * HD_ + t * 16 + l16];
      oacc[t] = wmma_f16(pa, vbf, oacc[t]);
    }
  }

#pragma unroll
  for (int r = 0; r < 8; ++r) {
    const float inv = 1.f / lrow[r];
    const int qidx = q0 + r + 8 * hlf;
#pragma unroll
    for (int t = 0; t < 4; ++t)
      ob[(size_t)qidx * HD_ + t * 16 + l16] = oacc[t][r] * inv;
  }
}

// ---------------------------------------------------------------------------
__global__ void ao_reorder(const float* __restrict__ attn,
                           _Float16* __restrict__ ao) {
  int idx = blockIdx.x * blockDim.x + threadIdx.x;
  if (idx >= B_ * NH_ * S_ * HD_) return;
  int d = idx & 63;
  int s = (idx >> 6) & (S_ - 1);
  int h = (idx >> 17) & (NH_ - 1);
  int b = idx >> 21;
  ao[((size_t)(b * S_ + s)) * H_ + h * HD_ + d] = (_Float16)attn[idx];
}

// ---------------------------------------------------------------------------
extern "C" void kernel_launch(void* const* d_in, const int* in_sizes, int n_in,
                              void* d_out, int out_size, void* d_ws,
                              size_t ws_size, hipStream_t stream) {
  const float* hidden = (const float*)d_in[0];
  const int*   pos    = (const int*)d_in[1];
  const float* q_w = (const float*)d_in[2];
  const float* q_b = (const float*)d_in[3];
  const float* k_w = (const float*)d_in[4];
  const float* k_b = (const float*)d_in[5];
  const float* v_w = (const float*)d_in[6];
  const float* v_b = (const float*)d_in[7];
  const float* o_w = (const float*)d_in[8];
  const float* lqa = (const float*)d_in[9];
  const float* lqb = (const float*)d_in[10];
  const float* lva = (const float*)d_in[11];
  const float* lvb = (const float*)d_in[12];
  const float* cdma = (const float*)d_in[13];
  float* out = (float*)d_out;

  char* ws = (char*)d_ws;
  size_t off = 0;
  auto alloc = [&](size_t bytes) {
    char* p = ws + off;
    off += (bytes + 255) & ~(size_t)255;
    return p;
  };

  const size_t NE = (size_t)B_ * NH_ * S_ * HD_;
  _Float16* Xh  = (_Float16*)alloc((size_t)M_ * H_ * 2);
  _Float16* Wqh = (_Float16*)alloc((size_t)H_ * H_ * 2);
  _Float16* Wkh = (_Float16*)alloc((size_t)H_ * H_ * 2);
  _Float16* Wvh = (_Float16*)alloc((size_t)H_ * H_ * 2);
  _Float16* Woh = (_Float16*)alloc((size_t)H_ * H_ * 2);
  float* qf = (float*)alloc((size_t)M_ * H_ * 4);
  float* kf = (float*)alloc((size_t)M_ * H_ * 4);
  float* vf = (float*)alloc((size_t)M_ * H_ * 4);
  float* tq = (float*)alloc((size_t)M_ * R_ * 4);
  float* tv = (float*)alloc((size_t)M_ * R_ * 4);
  _Float16* qr  = (_Float16*)alloc(NE * 2);
  float*    krf = (float*)alloc(NE * 4);
  float*    vrf = (float*)alloc(NE * 4);
  _Float16* Kn  = (_Float16*)alloc(NE * 2);
  _Float16* Vn  = (_Float16*)alloc(NE * 2);
  float*    att = (float*)alloc(NE * 4);
  _Float16* aoh = (_Float16*)alloc(NE * 2);

  const int T = 256;
  conv_f32_to_f16<<<(M_ * H_ + T - 1) / T, T, 0, stream>>>(hidden, Xh, M_ * H_);
  conv_f32_to_f16<<<(H_ * H_ + T - 1) / T, T, 0, stream>>>(q_w, Wqh, H_ * H_);
  conv_f32_to_f16<<<(H_ * H_ + T - 1) / T, T, 0, stream>>>(k_w, Wkh, H_ * H_);
  conv_f32_to_f16<<<(H_ * H_ + T - 1) / T, T, 0, stream>>>(v_w, Wvh, H_ * H_);
  conv_f32_to_f16<<<(H_ * H_ + T - 1) / T, T, 0, stream>>>(o_w, Woh, H_ * H_);

  const dim3 gg(H_ / BN_, M_ / BM_);   // (16, 32)
  gemm_lds_wmma<<<gg, T, 0, stream>>>(Xh, Wqh, qf, M_, H_, H_);
  gemm_lds_wmma<<<gg, T, 0, stream>>>(Xh, Wkh, kf, M_, H_, H_);
  gemm_lds_wmma<<<gg, T, 0, stream>>>(Xh, Wvh, vf, M_, H_, H_);

  lora_down<<<(M_ * R_ + T - 1) / T, T, 0, stream>>>(hidden, lqa, tq);
  lora_down<<<(M_ * R_ + T - 1) / T, T, 0, stream>>>(hidden, lva, tv);

  add_bias_lora<<<(M_ * H_ + T - 1) / T, T, 0, stream>>>(qf, q_b, tq, lqb, 1);
  add_bias_lora<<<(M_ * H_ + T - 1) / T, T, 0, stream>>>(kf, k_b, tq, lqb, 0);
  add_bias_lora<<<(M_ * H_ + T - 1) / T, T, 0, stream>>>(vf, v_b, tv, lvb, 1);

  rope_reorder<<<(int)((NE + T - 1) / T), T, 0, stream>>>(qf, kf, vf, pos, qr,
                                                          krf, vrf);
  {
    int n = BH_ * (S_ / KF_) * HD_;
    holo_bind<<<(n + T - 1) / T, T, 0, stream>>>(krf, vrf, cdma, Kn, Vn);
  }

  attn_wmma<<<dim3(S_ / 128, BH_), T, 0, stream>>>(qr, Kn, Vn, att);

  ao_reorder<<<(int)((NE + T - 1) / T), T, 0, stream>>>(att, aoh);
  gemm_lds_wmma<<<gg, T, 0, stream>>>(aoh, Woh, out, M_, H_, H_);
}
